// RPN_67456756351307
// MI455X (gfx1250) — matrix operand order repaired and checked by reading, not compiled, not run on hardware
//
#include <hip/hip_runtime.h>
#include <math.h>

// ---------------------------------------------------------------------------
// RPN pipeline for MI455X (gfx1250, wave32, WMMA 16x16x32 f16)
// LDS-free GEMMs: WMMA fragments are loaded straight from global memory with
// aligned b128 loads, using pre-packed f16 operands:
//   k0: features f32 NCHW -> xh  f16 [b][p][c]      (pixel-major)
//   k1: w1   -> Wt1 f16 [co][k=tap*256+c]           (k-contiguous rows)
//   k2: heads-> Wt2 f16 [n<64][k=ci]                (18 cls + 36 reg + pad)
//   k3: conv3x3+bias+ReLU -> f_t16 [b][p][co] f16   (WMMA, no LDS)
//   k4: 1x1 heads -> cls (b,p,18), reg (b,p,36)     (WMMA, no LDS)
//   k5: bbox decode + per-batch NMS (scores in LDS)
// ---------------------------------------------------------------------------

typedef __attribute__((ext_vector_type(16))) _Float16 v16h;
typedef __attribute__((ext_vector_type(8)))  _Float16 v8h;
typedef __attribute__((ext_vector_type(8)))  float    v8f;

#define BATCH   32
#define CIN     256
#define COUT    256
#define HW      50
#define NPIX    2500
#define KTOT    2304      // 9 taps * 256 channels, tap-major
#define CLS_C   18
#define REG_C   36
#define HEAD_N  64        // 54 padded to 64
#define NMS_K   1000
#define NEG_INF (-__builtin_inff())

union V16 { v16h v; v8h h[2]; };

// ---------------- packing kernels ----------------
// features f32 NCHW -> f16 [b][p][c]; p-fastest gid so global reads coalesce,
// scattered f16 writes are absorbed by the 192MB L2 (one-shot).
__global__ void rpn_pack_x(const float* __restrict__ x, _Float16* __restrict__ xh) {
  int gid = blockIdx.x * blockDim.x + threadIdx.x;   // [0, 32*256*2500)
  if (gid >= BATCH * CIN * NPIX) return;
  int p  = gid % NPIX;
  int r  = gid / NPIX;
  int c  = r & 255;
  int b  = r >> 8;
  xh[((size_t)b * NPIX + p) * CIN + c] = (_Float16)x[gid];  // x[(b*256+c)*2500+p]
}

__global__ void rpn_pack_w1(const float* __restrict__ w1, _Float16* __restrict__ Wt1) {
  int gid = blockIdx.x * blockDim.x + threadIdx.x;   // [0, 256*2304)
  if (gid >= COUT * KTOT) return;
  int k   = gid % KTOT;
  int co  = gid / KTOT;
  int tap = k >> 8;                                  // 0..8 = ky*3+kx
  int c   = k & 255;
  Wt1[gid] = (_Float16)w1[(co * CIN + c) * 9 + tap]; // Wt1[co*2304+k]
}

__global__ void rpn_pack_head(const float* __restrict__ wcls, const float* __restrict__ wreg,
                              _Float16* __restrict__ Wt2) {
  int gid = blockIdx.x * blockDim.x + threadIdx.x;   // [0, 64*256)
  if (gid >= HEAD_N * CIN) return;
  int k = gid & 255;
  int n = gid >> 8;
  float v = 0.f;
  if (n < CLS_C)               v = wcls[n * CIN + k];
  else if (n < CLS_C + REG_C)  v = wreg[(n - CLS_C) * CIN + k];
  Wt2[gid] = (_Float16)v;                            // Wt2[n*256+k]
}

// ---------------- conv 3x3 + bias + ReLU via WMMA (LDS-free) ----------------
// grid (20 pixel-tiles(128), 4 co-tiles(64), 32 batch), block 256 = 8 waves.
// Each wave: 16 px x 64 co, 4 accumulators. K = 2304 in 72 steps of 32.
__global__ __launch_bounds__(256) void rpn_conv3_wmma(
    const _Float16* __restrict__ xh, const _Float16* __restrict__ Wt1,
    const float* __restrict__ bias1, _Float16* __restrict__ f_t) {
  const int tid  = threadIdx.x;
  const int lane = tid & 31;
  const int wave = tid >> 5;
  const int m16  = lane & 15;
  const int kh   = lane >> 4;
  const int ptile = blockIdx.x * 128;
  const int co0   = blockIdx.y * 64;
  const int b     = blockIdx.z;

  const int pm = ptile + wave * 16 + m16;     // this lane's A-row pixel
  const int py = pm / HW;
  const int px = pm % HW;

  v8f acc0 = {}, acc1 = {}, acc2 = {}, acc3 = {};

  // B row pointers: n = co0 + j*16 + m16, K run starts at kh*16 (32B aligned)
  const _Float16* bp0 = Wt1 + (size_t)(co0 + 0  + m16) * KTOT + kh * 16;
  const _Float16* bp1 = Wt1 + (size_t)(co0 + 16 + m16) * KTOT + kh * 16;
  const _Float16* bp2 = Wt1 + (size_t)(co0 + 32 + m16) * KTOT + kh * 16;
  const _Float16* bp3 = Wt1 + (size_t)(co0 + 48 + m16) * KTOT + kh * 16;
  const _Float16* xb  = xh + (size_t)b * NPIX * CIN;

  const v8h az = {};

  for (int tap = 0; tap < 9; ++tap) {
    const int yy = py + tap / 3 - 1;
    const int xx = px + tap % 3 - 1;
    const bool valid = (pm < NPIX) & (yy >= 0) & (yy < HW) & (xx >= 0) & (xx < HW);
    const _Float16* ap = xb + ((size_t)(yy * HW + xx)) * CIN + kh * 16;
    const int kt = tap * 256;
    #pragma unroll
    for (int cb = 0; cb < 8; ++cb) {
      V16 a;
      // A frag: lane (m16,kh) holds K = kh*16+[0,16)
      a.h[0] = valid ? *(const v8h*)(ap + cb * 32)      : az;
      a.h[1] = valid ? *(const v8h*)(ap + cb * 32 + 8) : az;
      const int ko = kt + cb * 32;
      V16 b0, b1, b2, b3;
      b0.v = *(const v16h*)(bp0 + ko);
      b1.v = *(const v16h*)(bp1 + ko);
      b2.v = *(const v16h*)(bp2 + ko);
      b3.v = *(const v16h*)(bp3 + ko);
      acc0 = __builtin_amdgcn_wmma_f32_16x16x32_f16(false, a.v, false, b0.v, (short)0, acc0, false, false);
      acc1 = __builtin_amdgcn_wmma_f32_16x16x32_f16(false, a.v, false, b1.v, (short)0, acc1, false, false);
      acc2 = __builtin_amdgcn_wmma_f32_16x16x32_f16(false, a.v, false, b2.v, (short)0, acc2, false, false);
      acc3 = __builtin_amdgcn_wmma_f32_16x16x32_f16(false, a.v, false, b3.v, (short)0, acc3, false, false);
    }
  }

  // C layout: VGPR r -> pixel = prow + r, channel = co0 + j*16 + m16
  const int prow = ptile + wave * 16 + kh * 8;
  const int c0 = co0 + m16;
  const float bb0 = bias1[c0], bb1 = bias1[c0 + 16], bb2 = bias1[c0 + 32], bb3 = bias1[c0 + 48];
  #pragma unroll
  for (int r = 0; r < 8; ++r) {
    int pp = prow + r;
    if (pp < NPIX) {
      _Float16* o = f_t + ((size_t)b * NPIX + pp) * COUT + c0;
      o[0]  = (_Float16)fmaxf(acc0[r] + bb0, 0.f);
      o[16] = (_Float16)fmaxf(acc1[r] + bb1, 0.f);
      o[32] = (_Float16)fmaxf(acc2[r] + bb2, 0.f);
      o[48] = (_Float16)fmaxf(acc3[r] + bb3, 0.f);
    }
  }
}

// ---------------- 1x1 cls/reg heads via WMMA (LDS-free) ----------------
// grid (20 pixel-tiles(128), 32 batch), block 256. N=64, K=256 in 8 steps.
__global__ __launch_bounds__(256) void rpn_head_wmma(
    const _Float16* __restrict__ f_t, const _Float16* __restrict__ Wt2,
    const float* __restrict__ bcls, const float* __restrict__ breg,
    float* __restrict__ out_cls, float* __restrict__ out_reg) {
  const int tid  = threadIdx.x;
  const int lane = tid & 31;
  const int wave = tid >> 5;
  const int m16  = lane & 15;
  const int kh   = lane >> 4;
  const int ptile = blockIdx.x * 128;
  const int b     = blockIdx.y;

  const int pm = ptile + wave * 16 + m16;
  const bool pvalid = pm < NPIX;

  v8f acc0 = {}, acc1 = {}, acc2 = {}, acc3 = {};

  const _Float16* ap  = f_t + ((size_t)b * NPIX + (pvalid ? pm : 0)) * COUT + kh * 16;
  const _Float16* bp0 = Wt2 + (size_t)(0  + m16) * CIN + kh * 16;
  const _Float16* bp1 = Wt2 + (size_t)(16 + m16) * CIN + kh * 16;
  const _Float16* bp2 = Wt2 + (size_t)(32 + m16) * CIN + kh * 16;
  const _Float16* bp3 = Wt2 + (size_t)(48 + m16) * CIN + kh * 16;

  const v8h az = {};

  #pragma unroll
  for (int cb = 0; cb < 8; ++cb) {
    V16 a;
    a.h[0] = pvalid ? *(const v8h*)(ap + cb * 32)      : az;
    a.h[1] = pvalid ? *(const v8h*)(ap + cb * 32 + 8) : az;
    const int ko = cb * 32;
    V16 b0, b1, b2, b3;
    b0.v = *(const v16h*)(bp0 + ko);
    b1.v = *(const v16h*)(bp1 + ko);
    b2.v = *(const v16h*)(bp2 + ko);
    b3.v = *(const v16h*)(bp3 + ko);
    acc0 = __builtin_amdgcn_wmma_f32_16x16x32_f16(false, a.v, false, b0.v, (short)0, acc0, false, false);
    acc1 = __builtin_amdgcn_wmma_f32_16x16x32_f16(false, a.v, false, b1.v, (short)0, acc1, false, false);
    acc2 = __builtin_amdgcn_wmma_f32_16x16x32_f16(false, a.v, false, b2.v, (short)0, acc2, false, false);
    acc3 = __builtin_amdgcn_wmma_f32_16x16x32_f16(false, a.v, false, b3.v, (short)0, acc3, false, false);
  }

  const int prow = ptile + wave * 16 + kh * 8;
  #pragma unroll
  for (int r = 0; r < 8; ++r) {
    int pp = prow + r;
    if (pp >= NPIX) continue;
    #pragma unroll
    for (int j = 0; j < 4; ++j) {
      int n = j * 16 + m16;
      float val = (j == 0) ? acc0[r] : (j == 1) ? acc1[r] : (j == 2) ? acc2[r] : acc3[r];
      if (n < CLS_C) {
        out_cls[(size_t)b * (NPIX * CLS_C) + pp * CLS_C + n] = val + bcls[n];
      } else if (n < CLS_C + REG_C) {
        int nr = n - CLS_C;
        out_reg[(size_t)b * (NPIX * REG_C) + pp * REG_C + nr] = val + breg[nr];
      }
    }
  }
}

// ---------------- bbox decode + NMS (one block per batch) ----------------
__global__ __launch_bounds__(1024) void rpn_nms(
    const float* __restrict__ anchors, const int* __restrict__ aidx,
    const float* __restrict__ out_cls, const float* __restrict__ out_reg,
    float* __restrict__ bws, float* __restrict__ out_prop,
    float* __restrict__ out_valid, int N) {
  extern __shared__ float sm[];                 // N scores in LDS
  __shared__ float red_v[32];
  __shared__ int   red_i[32];
  __shared__ float s_box[4];
  __shared__ float s_v;
  __shared__ int   s_j;

  const int b    = blockIdx.x;
  const int tid  = threadIdx.x;
  const int lane = tid & 31;
  const int wave = tid >> 5;

  // Phase 1: decode boxes, compute scores
  for (int i = tid; i < N; i += 1024) {
    int ai = aidx[i];
    float ax1 = anchors[i * 4 + 0], ay1 = anchors[i * 4 + 1];
    float ax2 = anchors[i * 4 + 2], ay2 = anchors[i * 4 + 3];
    const float* d = out_reg + (size_t)b * (NPIX * REG_C) + ai * 4;
    float dx = d[0], dy = d[1], dw = d[2], dh = d[3];
    float w  = ax2 - ax1 + 1.f, h = ay2 - ay1 + 1.f;
    float cx = ax1 + 0.5f * w,  cy = ay1 + 0.5f * h;
    float pcx = dx * w + cx,    pcy = dy * h + cy;
    float pw  = expf(dw) * w,   ph  = expf(dh) * h;
    float x1 = fmaxf(pcx - 0.5f * pw, 0.f);
    float y1 = fmaxf(pcy - 0.5f * ph, 0.f);
    float x2 = fmaxf(pcx + 0.5f * pw, 0.f);
    float y2 = fmaxf(pcy + 0.5f * ph, 0.f);
    float* bp = bws + (size_t)(b * N + i) * 4;
    bp[0] = x1; bp[1] = y1; bp[2] = x2; bp[3] = y2;
    float cx1 = fminf(x1, 800.f), cx2 = fminf(x2, 800.f);
    float cy1 = fminf(y1, 800.f), cy2 = fminf(y2, 800.f);
    bool valid = (cx2 - cx1 >= 10.f) && (cy2 - cy1 >= 10.f);
    sm[i] = valid ? out_cls[(size_t)b * (NPIX * CLS_C) + ai * 2 + 1] : NEG_INF;
  }
  __syncthreads();

  // Phase 2: sequential greedy NMS, k = 1000 (argmax prefers lowest index)
  for (int it = 0; it < NMS_K; ++it) {
    float bv = NEG_INF;
    int   bi = 0x7fffffff;
    for (int i = tid; i < N; i += 1024) {
      float v = sm[i];
      if (v > bv || (v == bv && i < bi)) { bv = v; bi = i; }
    }
    #pragma unroll
    for (int off = 16; off >= 1; off >>= 1) {
      float ov = __shfl_down(bv, off);
      int   oi = __shfl_down(bi, off);
      if (ov > bv || (ov == bv && oi < bi)) { bv = ov; bi = oi; }
    }
    if (lane == 0) { red_v[wave] = bv; red_i[wave] = bi; }
    __syncthreads();
    if (tid < 32) {
      bv = red_v[tid]; bi = red_i[tid];
      #pragma unroll
      for (int off = 16; off >= 1; off >>= 1) {
        float ov = __shfl_down(bv, off);
        int   oi = __shfl_down(bi, off);
        if (ov > bv || (ov == bv && oi < bi)) { bv = ov; bi = oi; }
      }
      if (tid == 0) {
        s_v = bv; s_j = bi;
        if (bv > NEG_INF) {
          const float* bp = bws + (size_t)(b * N + bi) * 4;
          s_box[0] = bp[0]; s_box[1] = bp[1]; s_box[2] = bp[2]; s_box[3] = bp[3];
        }
      }
    }
    __syncthreads();

    if (!(s_v > NEG_INF)) {               // nothing left: zero-fill the tail
      for (int jj = it + tid; jj < NMS_K; jj += 1024) {
        float* op = out_prop + (size_t)(b * NMS_K + jj) * 4;
        op[0] = 0.f; op[1] = 0.f; op[2] = 0.f; op[3] = 0.f;
        out_valid[b * NMS_K + jj] = 0.f;
      }
      break;
    }
    const int   j   = s_j;
    const float jx1 = s_box[0], jy1 = s_box[1], jx2 = s_box[2], jy2 = s_box[3];
    const float areaj = (jx2 - jx1) * (jy2 - jy1);
    if (tid == 0) {
      float* op = out_prop + (size_t)(b * NMS_K + it) * 4;
      op[0] = jx1; op[1] = jy1; op[2] = jx2; op[3] = jy2;
      out_valid[b * NMS_K + it] = 1.f;
    }
    for (int i = tid; i < N; i += 1024) {
      float v = sm[i];
      if (v == NEG_INF) continue;         // already suppressed / invalid
      const float* bp = bws + (size_t)(b * N + i) * 4;
      float x1 = bp[0], y1 = bp[1], x2 = bp[2], y2 = bp[3];
      float lx = fmaxf(x1, jx1), ly = fmaxf(y1, jy1);
      float rx = fminf(x2, jx2), ry = fminf(y2, jy2);
      float iw = fmaxf(rx - lx, 0.f), ih = fmaxf(ry - ly, 0.f);
      float inter = iw * ih;
      float ar    = (x2 - x1) * (y2 - y1);
      float iou   = inter / (ar + areaj - inter);
      if (iou > 0.7f || i == j) sm[i] = NEG_INF;
    }
    __syncthreads();
  }
}

// ---------------------------------------------------------------------------
extern "C" void kernel_launch(void* const* d_in, const int* in_sizes, int n_in,
                              void* d_out, int out_size, void* d_ws, size_t ws_size,
                              hipStream_t stream) {
  const float* features = (const float*)d_in[0];
  const float* anchors  = (const float*)d_in[1];
  const int*   aidx     = (const int*)  d_in[2];
  const float* w1       = (const float*)d_in[3];
  const float* b1       = (const float*)d_in[4];
  const float* w_cls    = (const float*)d_in[5];
  const float* b_cls    = (const float*)d_in[6];
  const float* w_reg    = (const float*)d_in[7];
  const float* b_reg    = (const float*)d_in[8];
  const int N = in_sizes[2];            // number of kept anchors

  // workspace layout
  char* ws = (char*)d_ws;
  _Float16* xh  = (_Float16*)ws;                                 // 32*2500*256 f16
  size_t off = (size_t)BATCH * NPIX * CIN * sizeof(_Float16);    // 40,960,000
  _Float16* f_t = (_Float16*)(ws + off);                         // 32*2500*256 f16
  off += (size_t)BATCH * NPIX * COUT * sizeof(_Float16);         // +40,960,000
  _Float16* Wt1 = (_Float16*)(ws + off);                         // 256*2304 f16
  off += (size_t)COUT * KTOT * sizeof(_Float16);                 // +1,179,648
  _Float16* Wt2 = (_Float16*)(ws + off);                         // 64*256 f16
  off += (size_t)HEAD_N * CIN * sizeof(_Float16);                // +32,768
  float* bws = (float*)(ws + off);                               // 32*N*4 f32 boxes

  // output layout (reference tuple order, flat)
  float* out_cls   = (float*)d_out;                              // 32*22500*2
  float* out_reg   = out_cls + (size_t)BATCH * NPIX * CLS_C;     // 32*22500*4
  float* out_prop  = out_cls + 4320000;                          // 32*1000*4
  float* out_valid = out_cls + 4448000;                          // 32*1000

  rpn_pack_x<<<(BATCH * CIN * NPIX + 255) / 256, 256, 0, stream>>>(features, xh);
  rpn_pack_w1<<<(COUT * KTOT + 255) / 256, 256, 0, stream>>>(w1, Wt1);
  rpn_pack_head<<<(HEAD_N * CIN + 255) / 256, 256, 0, stream>>>(w_cls, w_reg, Wt2);

  dim3 gconv((NPIX + 127) / 128, COUT / 64, BATCH);   // 20 x 4 x 32
  rpn_conv3_wmma<<<gconv, 256, 0, stream>>>(xh, Wt1, b1, f_t);

  dim3 ghead((NPIX + 127) / 128, BATCH, 1);           // 20 x 32
  rpn_head_wmma<<<ghead, 256, 0, stream>>>(f_t, Wt2, b_cls, b_reg, out_cls, out_reg);

  size_t smbytes = (size_t)N * sizeof(float);         // scores in LDS (~90 KB)
  rpn_nms<<<BATCH, 1024, smbytes, stream>>>(anchors, aidx, out_cls, out_reg,
                                            bws, out_prop, out_valid, N);
}
